// CausalSelfAttention_9809705305052
// MI455X (gfx1250) — matrix operand order repaired and checked by reading, not compiled, not run on hardware
//
#include <hip/hip_runtime.h>
#include <hip/hip_bf16.h>

// ---------- types ----------
typedef __bf16 bf16;
typedef bf16  bf16x16 __attribute__((ext_vector_type(16)));
typedef bf16  bf16x8  __attribute__((ext_vector_type(8)));
typedef float f32x8   __attribute__((ext_vector_type(8)));
typedef unsigned int u32;
typedef u32 u32x4 __attribute__((ext_vector_type(4)));
typedef int  i32x4 __attribute__((ext_vector_type(4)));
typedef int  i32x8 __attribute__((ext_vector_type(8)));

// A-matrix 16x32 bf16 fragment layout (per ISA 7.12.2):
// lanes 0-15 (M=lane): elems 0..7 -> K=0..7,  elems 8..15 -> K=16..23
// lanes16-31 (M=lane-16): elems 0..7 -> K=8..15, elems 8..15 -> K=24..31
__device__ __forceinline__ bf16x16 load_afrag(const bf16* p, int hw) {
  bf16x8 lo = *(const bf16x8*)(p + hw * 8);
  bf16x8 hi = *(const bf16x8*)(p + 16 + hw * 8);
  bf16x16 r;
#pragma unroll
  for (int i = 0; i < 8; ++i) { r[i] = lo[i]; r[i + 8] = hi[i]; }
  return r;
}

__device__ __forceinline__ f32x8 wmma_bf16(bf16x16 a, bf16x16 b, f32x8 c) {
  return __builtin_amdgcn_wmma_f32_16x16x32_bf16(false, a, false, b, (short)0, c,
                                                 false, false);
}

// LDS byte offset of a __shared__ object (generic -> AS(3) -> int)
__device__ __forceinline__ u32 lds_off(const void* p) {
  return (u32)(uintptr_t)(__attribute__((address_space(3))) const void*)p;
}

// ---------- Tensor Data Mover: 2D tile load Global -> LDS ----------
// D# per ISA ch.8: group0 {count=1, lds_addr, global_addr[56:0], type=2},
// group1 {data_size=2B, tensor_dim0/1, tile_dim0/1, tensor_dim0_stride}.
__device__ __forceinline__ void tdm_load_2d(const void* gaddr, u32 ldsoff,
                                            u32 dim0, u32 dim1, u32 stride0,
                                            u32 tile0, u32 tile1) {
  unsigned long long ga = (unsigned long long)(uintptr_t)gaddr;
  u32x4 g0;
  g0[0] = 1u;                                        // count=1 (valid descriptor)
  g0[1] = ldsoff;                                    // lds_addr [63:32]
  g0[2] = (u32)(ga & 0xffffffffu);                   // global_addr lo
  g0[3] = (u32)((ga >> 32) & 0x1ffffffu) | (2u << 30);  // ga hi | type=2
  i32x8 g1;
  g1[0] = (int)(1u << 16);                           // workgroup_mask=0, data_size=1 (2B)
  g1[1] = (int)((dim0 & 0xffffu) << 16);             // tensor_dim0[15:0] -> [63:48]
  g1[2] = (int)(((dim0 >> 16) & 0xffffu) | ((dim1 & 0xffffu) << 16));
  g1[3] = (int)(((dim1 >> 16) & 0xffffu) | ((tile0 & 0xffffu) << 16)); // tile_dim0
  g1[4] = (int)(tile1 & 0xffffu);                    // tile_dim1 (tile_dim2=0)
  g1[5] = (int)stride0;                              // tensor_dim0_stride[31:0]
  g1[6] = 0;                                         // stride0 hi, stride1 lo
  g1[7] = 0;
  i32x4 z = {};
#if defined(__clang_major__) && (__clang_major__ >= 23)
  i32x8 z8 = {};
  __builtin_amdgcn_tensor_load_to_lds(g0, g1, z, z, z8, 0);
#else
  __builtin_amdgcn_tensor_load_to_lds(g0, g1, z, z, 0);
#endif
}

// ---------- fp32 -> bf16 conversion ----------
__global__ __launch_bounds__(256) void conv_f32_bf16(const float* __restrict__ in,
                                                     bf16* __restrict__ out, int n) {
  int i = (blockIdx.x * 256 + threadIdx.x) * 8;
  if (i + 8 <= n) {
    f32x8 v = *(const f32x8*)(in + i);
    bf16x8 o;
#pragma unroll
    for (int j = 0; j < 8; ++j) o[j] = (bf16)v[j];
    *(bf16x8*)(out + i) = o;
  }
}

// ---------- WMMA GEMM with TDM double-buffered LDS staging ----------
// C[M,N] = A[M,K] @ B[N,K]^T. block = 256 thr = 8 waves (4x2); block tile 128x64;
// wave tile 32x32. Per 32-wide K slab, TDM streams A(128x32) and B(64x32) to LDS.
__global__ __launch_bounds__(256) void gemm_bf16_nt(const bf16* __restrict__ A,
                                                    const bf16* __restrict__ Bm,
                                                    float* __restrict__ C,
                                                    int M, int N, int K) {
  __shared__ __align__(64) bf16 Ash[2][128][32];
  __shared__ __align__(64) bf16 Bsh[2][64][32];
  const int lane = threadIdx.x & 31;
  const int wave = threadIdx.x >> 5;
  const int wm = wave >> 1, wn = wave & 1;
  const int l15 = lane & 15, hw = lane >> 4;
  const int m0 = blockIdx.y * 128;
  const int n0 = blockIdx.x * 64;
  const int nslab = K >> 5;
  const bool issuer = (threadIdx.x < 32);   // wave 0 drives the TDM

  if (issuer) {
    tdm_load_2d(A + (size_t)m0 * K, lds_off(&Ash[0][0][0]), K, M, K, 32, 128);
    tdm_load_2d(Bm + (size_t)n0 * K, lds_off(&Bsh[0][0][0]), K, N, K, 32, 64);
  }

  f32x8 acc[2][2] = {};
  for (int ks = 0; ks < nslab; ++ks) {
    const int buf = ks & 1;
    if (issuer) __builtin_amdgcn_s_wait_tensorcnt(0);  // slab ks landed in LDS
    __syncthreads();  // data visible to all waves; prior slab fully consumed
    if (issuer && (ks + 1 < nslab)) {                  // prefetch next slab (async DMA)
      const int k1 = (ks + 1) * 32;
      tdm_load_2d(A + (size_t)m0 * K + k1, lds_off(&Ash[buf ^ 1][0][0]), K, M, K, 32, 128);
      tdm_load_2d(Bm + (size_t)n0 * K + k1, lds_off(&Bsh[buf ^ 1][0][0]), K, N, K, 32, 64);
    }
    bf16x16 a0 = load_afrag(&Ash[buf][wm * 32 + l15][0], hw);
    bf16x16 a1 = load_afrag(&Ash[buf][wm * 32 + 16 + l15][0], hw);
    bf16x16 b0 = load_afrag(&Bsh[buf][wn * 32 + l15][0], hw);
    bf16x16 b1 = load_afrag(&Bsh[buf][wn * 32 + 16 + l15][0], hw);
    acc[0][0] = wmma_bf16(a0, b0, acc[0][0]);
    acc[0][1] = wmma_bf16(a0, b1, acc[0][1]);
    acc[1][0] = wmma_bf16(a1, b0, acc[1][0]);
    acc[1][1] = wmma_bf16(a1, b1, acc[1][1]);
  }
  // C/D layout: lane -> col (l15), VGPR r -> row r + 8*hw
#pragma unroll
  for (int mi = 0; mi < 2; ++mi)
#pragma unroll
    for (int ni = 0; ni < 2; ++ni)
#pragma unroll
      for (int r = 0; r < 8; ++r) {
        int m = m0 + wm * 32 + mi * 16 + r + 8 * hw;
        int n = n0 + wn * 32 + ni * 16 + l15;
        C[(size_t)m * N + n] = acc[mi][ni][r];
      }
}

// ---------- QKV epilogue: RMSNorm + partial RoPE + gain; layout rewrite ----------
// qkv f32 [B*T][1536] : q cols 0..1023 (h*64+d), k cols 1024..1279, v cols 1280..1535
// Q -> bf16 [B][16][T][64], K -> bf16 [B][4][T][64], Vt -> bf16 [B][4][64][T]
__global__ __launch_bounds__(256) void qkv_prep(const float* __restrict__ qkv,
                                                const float* __restrict__ q_gain,
                                                bf16* __restrict__ Qb,
                                                bf16* __restrict__ Kb,
                                                bf16* __restrict__ Vt) {
  const int row = blockIdx.x;         // b*T + t
  const int b = row >> 11;            // T = 2048
  const int t = row & 2047;
  const float* rp = qkv + (size_t)row * 1536;
  const int tid = threadIdx.x;
  const int grp = tid >> 4;           // 16-thread group id
  const int l16 = tid & 15;
  const int d0 = l16 * 4;
  const float eps = 1.1920929e-7f;
  const float kLog10k_32 = 9.2103403720f / 32.0f;  // ln(10000)/32
  const float tf = (float)t;

  // ---- Q: group = head h (0..15), 4 dims per thread ----
  {
    const int h = grp;
    float v[4];
#pragma unroll
    for (int j = 0; j < 4; ++j) v[j] = rp[h * 64 + d0 + j];
    float ss = v[0] * v[0] + v[1] * v[1] + v[2] * v[2] + v[3] * v[3];
#pragma unroll
    for (int off = 1; off < 16; off <<= 1) ss += __shfl_xor(ss, off);
    const float rms = rsqrtf(ss * (1.0f / 64.0f) + eps);
#pragma unroll
    for (int j = 0; j < 4; ++j) v[j] *= rms;
    float part[4];
#pragma unroll
    for (int j = 0; j < 4; ++j) part[j] = __shfl_xor(v[j], 2);
    if (d0 < 16) {
#pragma unroll
      for (int j = 0; j < 4; ++j) {
        const int d = d0 + j;
        const int i = (d < 8) ? d : d - 8;
        const float ang = tf * __expf(-(float)i * kLog10k_32);
        const float c = __cosf(ang), s = __sinf(ang);
        v[j] = (d < 8) ? (v[j] * c + part[j] * s) : (-part[j] * s + v[j] * c);
      }
    }
    const float g = q_gain[h];
    bf16* qo = Qb + (((size_t)(b * 16 + h)) * 2048 + t) * 64 + d0;
#pragma unroll
    for (int j = 0; j < 4; ++j) qo[j] = (bf16)(v[j] * g);
  }

  // ---- K: threads 0..63, group = kv head (0..3) ----
  if (tid < 64) {
    const int hk = grp;
    float v[4];
#pragma unroll
    for (int j = 0; j < 4; ++j) v[j] = rp[1024 + hk * 64 + d0 + j];
    float ss = v[0] * v[0] + v[1] * v[1] + v[2] * v[2] + v[3] * v[3];
#pragma unroll
    for (int off = 1; off < 16; off <<= 1) ss += __shfl_xor(ss, off);
    const float rms = rsqrtf(ss * (1.0f / 64.0f) + eps);
#pragma unroll
    for (int j = 0; j < 4; ++j) v[j] *= rms;
    float part[4];
#pragma unroll
    for (int j = 0; j < 4; ++j) part[j] = __shfl_xor(v[j], 2);
    if (d0 < 16) {
#pragma unroll
      for (int j = 0; j < 4; ++j) {
        const int d = d0 + j;
        const int i = (d < 8) ? d : d - 8;
        const float ang = tf * __expf(-(float)i * kLog10k_32);
        const float c = __cosf(ang), s = __sinf(ang);
        v[j] = (d < 8) ? (v[j] * c + part[j] * s) : (-part[j] * s + v[j] * c);
      }
    }
    bf16* ko = Kb + (((size_t)(b * 4 + hk)) * 2048 + t) * 64 + d0;
#pragma unroll
    for (int j = 0; j < 4; ++j) ko[j] = (bf16)v[j];
  }

  // ---- V: transpose to [b][hk][d][t] ----
  {
    const int c = tid;                // 0..255 == hk*64 + d
    const int hk = c >> 6, d = c & 63;
    Vt[(((size_t)(b * 4 + hk)) * 64 + d) * 2048 + t] = (bf16)rp[1280 + c];
  }
}

// ---------- Flash attention: wave-per-16-q-rows, kv tiles of 32 ----------
// Q [B][16][T][64], K [B][4][T][64], Vt [B][4][64][T], out bf16 [B*T][1024]
__global__ __launch_bounds__(128) void attn_kernel(const bf16* __restrict__ Q,
                                                   const bf16* __restrict__ Kb,
                                                   const bf16* __restrict__ Vt,
                                                   const int* __restrict__ wlp,
                                                   bf16* __restrict__ Ob) {
  __shared__ __align__(32) bf16 pshm[4][16][32];   // per-wave P transpose buffer
  const int lane = threadIdx.x & 31;
  const int wave = threadIdx.x >> 5;
  const int l15 = lane & 15, hw = lane >> 4;
  const int bh = blockIdx.y;
  const int b = bh >> 4, h = bh & 15, hk = h >> 2;
  const int q0 = blockIdx.x * 64 + wave * 16;
  const int wl = *wlp;
  const float scale = 0.125f;                      // 1/sqrt(64)

  const bf16* qrow = Q + (((size_t)(b * 16 + h)) * 2048 + (q0 + l15)) * 64;
  bf16x16 qf0 = load_afrag(qrow, hw);
  bf16x16 qf1 = load_afrag(qrow + 32, hw);

  const bf16* Kbase = Kb + ((size_t)(b * 4 + hk)) * 2048 * 64;
  const bf16* Vbase = Vt + ((size_t)(b * 4 + hk)) * 64 * 2048;

  f32x8 o[4] = {};
  float mrun[8], lrun[8];
#pragma unroll
  for (int r = 0; r < 8; ++r) { mrun[r] = -1e30f; lrun[r] = 0.0f; }

  int kv_lo = 0;
  if (wl >= 0) { kv_lo = q0 - wl; if (kv_lo < 0) kv_lo = 0; kv_lo &= ~31; }

  for (int kv0 = kv_lo; kv0 <= q0 + 15; kv0 += 32) {
    // prefetch next kv tile (speculative; dropped if unused)
    __builtin_prefetch(Kbase + (size_t)(kv0 + 32 + l15) * 64, 0, 0);
    __builtin_prefetch(Vbase + (size_t)(lane * 2) * 2048 + kv0 + 32, 0, 0);
    // ---- S = Q K^T (two 16-col tiles) ----
    f32x8 s[2] = {};
#pragma unroll
    for (int ni = 0; ni < 2; ++ni) {
      const bf16* kr = Kbase + (size_t)(kv0 + ni * 16 + l15) * 64;
      bf16x16 b0 = load_afrag(kr, hw);
      bf16x16 b1 = load_afrag(kr + 32, hw);
      s[ni] = wmma_bf16(qf0, b0, s[ni]);
      s[ni] = wmma_bf16(qf1, b1, s[ni]);
    }
    // ---- scale + causal/window mask + row max ----
    float rm[8];
#pragma unroll
    for (int r = 0; r < 8; ++r) {
      const int qa = q0 + r + 8 * hw;
#pragma unroll
      for (int ni = 0; ni < 2; ++ni) {
        const int j = kv0 + ni * 16 + l15;
        float sv = s[ni][r] * scale;
        const bool ok = (j <= qa) && (wl < 0 || (qa - j) <= wl);
        s[ni][r] = ok ? sv : -1e30f;
      }
      rm[r] = fmaxf(s[0][r], s[1][r]);
    }
#pragma unroll
    for (int off = 1; off < 16; off <<= 1)
#pragma unroll
      for (int r = 0; r < 8; ++r) rm[r] = fmaxf(rm[r], __shfl_xor(rm[r], off));
    // ---- online softmax update ----
    float alpha[8], rs[8];
#pragma unroll
    for (int r = 0; r < 8; ++r) {
      const float mnew = fmaxf(mrun[r], rm[r]);
      alpha[r] = __expf(mrun[r] - mnew);
      const float s0 = s[0][r], s1 = s[1][r];
      const float p0 = (s0 <= -1e29f) ? 0.0f : __expf(s0 - mnew);
      const float p1 = (s1 <= -1e29f) ? 0.0f : __expf(s1 - mnew);
      s[0][r] = p0; s[1][r] = p1;
      rs[r] = p0 + p1;
      mrun[r] = mnew;
    }
#pragma unroll
    for (int off = 1; off < 16; off <<= 1)
#pragma unroll
      for (int r = 0; r < 8; ++r) rs[r] += __shfl_xor(rs[r], off);
#pragma unroll
    for (int r = 0; r < 8; ++r) lrun[r] = lrun[r] * alpha[r] + rs[r];
#pragma unroll
    for (int di = 0; di < 4; ++di)
#pragma unroll
      for (int r = 0; r < 8; ++r) o[di][r] *= alpha[r];
    // ---- transpose P (C-layout) -> A-layout via per-wave LDS ----
    asm volatile("s_wait_dscnt 0x0" ::: "memory");  // prior reads done
#pragma unroll
    for (int ni = 0; ni < 2; ++ni)
#pragma unroll
      for (int r = 0; r < 8; ++r)
        pshm[wave][r + 8 * hw][ni * 16 + l15] = (bf16)s[ni][r];
    asm volatile("s_wait_dscnt 0x0" ::: "memory");  // writes visible
    const bf16x16 pf = load_afrag(&pshm[wave][l15][0], hw);
    // ---- O += P V ----
#pragma unroll
    for (int di = 0; di < 4; ++di) {
      const bf16* vr = Vbase + (size_t)(di * 16 + l15) * 2048 + kv0;
      bf16x16 vf = load_afrag(vr, hw);
      o[di] = wmma_bf16(pf, vf, o[di]);
    }
  }
  // ---- normalize + store bf16 into [b*T+t][h*64+d] ----
#pragma unroll
  for (int r = 0; r < 8; ++r) {
    const float inv = 1.0f / lrun[r];
    const size_t trow = (size_t)b * 2048 + q0 + r + 8 * hw;
#pragma unroll
    for (int di = 0; di < 4; ++di)
      Ob[trow * 1024 + h * 64 + di * 16 + l15] = (bf16)(o[di][r] * inv);
  }
}

// ---------- host launcher ----------
extern "C" void kernel_launch(void* const* d_in, const int* in_sizes, int n_in,
                              void* d_out, int out_size, void* d_ws, size_t ws_size,
                              hipStream_t stream) {
  const float* x      = (const float*)d_in[0];   // [2,2048,1024]
  const float* wq     = (const float*)d_in[1];   // [1024,1024]
  const float* wk     = (const float*)d_in[2];   // [256,1024]
  const float* wv     = (const float*)d_in[3];   // [256,1024]
  const float* wproj  = (const float*)d_in[4];   // [1024,1024]
  const float* q_gain = (const float*)d_in[5];   // [16]
  const int*   wl     = (const int*)d_in[6];     // scalar

  char* ws = (char*)d_ws;
  bf16*  xbf  = (bf16*)(ws + 0);                 //  8 MB  [4096][1024]
  bf16*  wqkv = (bf16*)(ws + 8388608);           //  3 MB  [1536][1024]
  bf16*  wpbf = (bf16*)(ws + 11534336);          //  2 MB  [1024][1024]
  float* qkv  = (float*)(ws + 13631488);         // 24 MB  [4096][1536]
  bf16*  Qb   = (bf16*)(ws + 38797312);          //  8 MB  [2][16][2048][64]
  bf16*  Kb   = (bf16*)(ws + 47185920);          //  2 MB  [2][4][2048][64]
  bf16*  Vt   = (bf16*)(ws + 49283072);          //  2 MB  [2][4][64][2048]
  bf16*  Ob   = (bf16*)(ws + 51380224);          //  8 MB  [4096][1024]

  conv_f32_bf16<<<2048, 256, 0, stream>>>(x, xbf, 4194304);
  conv_f32_bf16<<<512,  256, 0, stream>>>(wq, wqkv, 1048576);
  conv_f32_bf16<<<128,  256, 0, stream>>>(wk, wqkv + 1048576, 262144);
  conv_f32_bf16<<<128,  256, 0, stream>>>(wv, wqkv + 1310720, 262144);
  conv_f32_bf16<<<512,  256, 0, stream>>>(wproj, wpbf, 1048576);

  gemm_bf16_nt<<<dim3(24, 32), 256, 0, stream>>>(xbf, wqkv, qkv, 4096, 1536, 1024);
  qkv_prep<<<4096, 256, 0, stream>>>(qkv, q_gain, Qb, Kb, Vt);
  attn_kernel<<<dim3(32, 32), 128, 0, stream>>>(Qb, Kb, Vt, wl, Ob);
  gemm_bf16_nt<<<dim3(16, 32), 256, 0, stream>>>(Ob, wpbf, (float*)d_out, 4096, 1024, 1024);
}
